// ConditionalMMDecoder_85968065397345
// MI455X (gfx1250) — matrix operand order repaired and checked by reading, not compiled, not run on hardware
//
#include <hip/hip_runtime.h>
#include <hip/hip_bf16.h>

// ---------------- problem constants (match reference) ----------------
#define HID   512
#define EMB   512
#define CTXD  512
#define NVOC  32000
#define BATCH 64
#define TLEN  48
#define STXT  50
#define SIMG  196
#define NEGBIG (-1e9f)
#define MT 4            // M tiles per wave in vocab GEMM (covers all BATCH=64)
#define NT 4            // N tiles per wave in vocab GEMM

typedef __attribute__((ext_vector_type(2)))  float  v2f;
typedef __attribute__((ext_vector_type(8)))  float  v8f;
typedef __attribute__((ext_vector_type(8)))  __bf16 v8bf;
typedef __attribute__((ext_vector_type(16))) __bf16 v16bf;

// =====================================================================
// Generic f32 WMMA GEMM:  D[M,N] = act( A[M,K] @ W[N,K]^T + bias[N] )
// grid = (N/32, M/16), block = 32: one wave per 16x32 (2 N tiles,
// shared A fragment, 2 interleaved accumulator chains).
// V_WMMA_F32_16X16X4_F32 layouts (ISA 7.12.2):
//   A 16x4: lanes 0-15 row M=lane, K=0,1; lanes 16-31 K=2,3
//   B 4x16: lanes 0-15 col N=lane, K=0,1; lanes 16-31 K=2,3
//   C/D:    VGPR r -> M = hh*8 + r, N = lane%16
// =====================================================================
__global__ __launch_bounds__(32)
void gemm_f32_wmma(const float* __restrict__ A,
                   const float* __restrict__ W,
                   const float* __restrict__ bias,
                   float* __restrict__ D,
                   int M, int N, int K, int act)
{
    int lane = threadIdx.x & 31;
    int rc   = lane & 15;        // row (A) / col (B)
    int hh   = lane >> 4;        // lane-group half
    int m0   = blockIdx.y << 4;
    int n0   = blockIdx.x << 5;  // 32 columns per wave

    const float* __restrict__ arow  = A + (size_t)(m0 + rc) * K;
    const float* __restrict__ brow0 = W + (size_t)(n0 + rc) * K;
    const float* __restrict__ brow1 = W + (size_t)(n0 + 16 + rc) * K;

    v8f acc0 = {}, acc1 = {};
    for (int k0 = 0; k0 < K; k0 += 4) {
        v2f a  = *(const v2f*)(arow  + k0 + hh * 2);
        v2f b0 = *(const v2f*)(brow0 + k0 + hh * 2);
        v2f b1 = *(const v2f*)(brow1 + k0 + hh * 2);
        acc0 = __builtin_amdgcn_wmma_f32_16x16x4_f32(
            false, a, false, b0, (short)0, acc0, false, false);
        acc1 = __builtin_amdgcn_wmma_f32_16x16x4_f32(
            false, a, false, b1, (short)0, acc1, false, false);
    }

    int col0 = n0 + rc, col1 = n0 + 16 + rc;
    float bv0 = bias ? bias[col0] : 0.0f;
    float bv1 = bias ? bias[col1] : 0.0f;
    #pragma unroll
    for (int r = 0; r < 8; ++r) {
        size_t rowoff = (size_t)(m0 + hh * 8 + r) * N;
        float v0 = acc0[r] + bv0;
        float v1 = acc1[r] + bv1;
        if (act) { v0 = tanhf(v0); v1 = tanhf(v1); }
        D[rowoff + col0] = v0;
        D[rowoff + col1] = v1;
    }
}

// =====================================================================
// Vocab GEMM (bf16 WMMA): D[BATCH,NVOC] = A[BATCH,EMB] @ Wbf[NVOC,EMB]^T + bias
// grid = (NVOC/(16*NT)) = 500 waves; each wave register-blocks the FULL
// batch (MT*16 = 64 rows) x 64 cols: 16 independent accumulator chains,
// B fragments reused across all 4 M tiles, A fragments across 4 N tiles.
// V_WMMA_F32_16X16X32_BF16 layouts (ISA 7.12.2, 16-bit):
//   A 16x32: lane L: row=L%16, hh=L/16; elem e -> K=(e<8?0:16)+hh*8+(e%8)
//   B 32x16: lane L: col=L%16, hh=L/16; elem e -> K=(e<8?0:16)+hh*8+(e%8)
// =====================================================================
__global__ __launch_bounds__(32)
void gemm_bf16_vocab(const __bf16* __restrict__ A,
                     const __bf16* __restrict__ W,
                     const float*  __restrict__ bias,
                     float* __restrict__ D)
{
    int lane = threadIdx.x & 31;
    int rc   = lane & 15;
    int hh   = lane >> 4;
    int n0   = blockIdx.x * (16 * NT);

    const __bf16* arow[MT];
    #pragma unroll
    for (int m = 0; m < MT; ++m)
        arow[m] = A + (size_t)(m * 16 + rc) * EMB;
    const __bf16* brow[NT];
    #pragma unroll
    for (int v = 0; v < NT; ++v)
        brow[v] = W + (size_t)(n0 + v * 16 + rc) * EMB;

    v8f acc[MT][NT] = {};
    for (int k0 = 0; k0 < EMB; k0 += 32) {
        // prefetch next K-chunk of the weight stream (global_prefetch_b8)
        if (k0 + 32 < EMB) __builtin_prefetch(brow[0] + k0 + 32, 0, 1);

        // A fragments: two contiguous 16B chunks each -> interleaved v16bf
        v16bf a[MT];
        #pragma unroll
        for (int m = 0; m < MT; ++m) {
            v8bf a_lo = *(const v8bf*)(arow[m] + k0 + hh * 8);
            v8bf a_hi = *(const v8bf*)(arow[m] + k0 + 16 + hh * 8);
            #pragma unroll
            for (int e = 0; e < 8; ++e) { a[m][e] = a_lo[e]; a[m][e + 8] = a_hi[e]; }
        }

        #pragma unroll
        for (int v = 0; v < NT; ++v) {
            v8bf b_lo = *(const v8bf*)(brow[v] + k0 + hh * 8);
            v8bf b_hi = *(const v8bf*)(brow[v] + k0 + 16 + hh * 8);
            v16bf b;
            #pragma unroll
            for (int e = 0; e < 8; ++e) { b[e] = b_lo[e]; b[e + 8] = b_hi[e]; }
            #pragma unroll
            for (int m = 0; m < MT; ++m)
                acc[m][v] = __builtin_amdgcn_wmma_f32_16x16x32_bf16(
                    false, a[m], false, b, (short)0, acc[m][v], false, false);
        }
    }

    #pragma unroll
    for (int v = 0; v < NT; ++v) {
        int col = n0 + v * 16 + rc;
        float bv = bias[col];
        #pragma unroll
        for (int m = 0; m < MT; ++m) {
            #pragma unroll
            for (int r = 0; r < 8; ++r)
                D[(size_t)(m * 16 + hh * 8 + r) * NVOC + col] = acc[m][v][r] + bv;
        }
    }
}

// =====================================================================
// Elementwise helpers
// =====================================================================
__device__ __forceinline__ float sigmoidf_(float x) { return 1.0f / (1.0f + expf(-x)); }

__global__ void gru_gates(const float* __restrict__ gi, const float* __restrict__ gh,
                          const float* __restrict__ h,  float* __restrict__ hout)
{
    int i = blockIdx.x * blockDim.x + threadIdx.x;
    if (i >= BATCH * HID) return;
    int b = i / HID, j = i - b * HID;
    const float* gib = gi + (size_t)b * (3 * HID);
    const float* ghb = gh + (size_t)b * (3 * HID);
    float r  = sigmoidf_(gib[j]       + ghb[j]);
    float z  = sigmoidf_(gib[HID + j] + ghb[HID + j]);
    float nn = tanhf(gib[2 * HID + j] + r * ghb[2 * HID + j]);
    hout[i] = (1.0f - z) * nn + z * h[i];
}

__global__ void embed_gather(const int* __restrict__ y, const float* __restrict__ embW,
                             float* __restrict__ yemb)
{
    int r = blockIdx.x;
    int tok = y[r];
    const float* src = embW + (size_t)tok * EMB;
    float* dst = yemb + (size_t)r * EMB;
    for (int c = threadIdx.x; c < EMB; c += blockDim.x)
        dst[c] = (tok == 0) ? 0.0f : src[c];
}

__global__ void f2bf_kernel(const float* __restrict__ x, __bf16* __restrict__ o, long n)
{
    long i = (long)blockIdx.x * blockDim.x + threadIdx.x;
    long stride = (long)gridDim.x * blockDim.x;
    for (; i < n; i += stride) o[i] = (__bf16)x[i];
}

__global__ void zero_f(float* __restrict__ p, int n)
{
    int i = blockIdx.x * blockDim.x + threadIdx.x;
    if (i < n) p[i] = 0.0f;
}

// =====================================================================
// MLP attention: one block per batch element (wave32 shuffle reductions)
// =====================================================================
__global__ void att_kernel(const float* __restrict__ proj, const float* __restrict__ ctx,
                           const float* __restrict__ hp,   const float* __restrict__ mlpv,
                           const float* __restrict__ mask, float* __restrict__ z,
                           int S, int zoff)
{
    __shared__ float sc[256];
    __shared__ float red[256];
    int b   = blockIdx.x;
    int tid = threadIdx.x;
    int wid = tid >> 5, lane = tid & 31;
    const float* hpb = hp + (size_t)b * CTXD;

    for (int s = wid; s < S; s += 8) {
        const float* pr = proj + ((size_t)s * BATCH + b) * CTXD;
        float acc = 0.0f;
        for (int c = lane; c < CTXD; c += 32)
            acc += tanhf(pr[c] + hpb[c]) * mlpv[c];
        for (int o = 16; o > 0; o >>= 1) acc += __shfl_down(acc, o, 32);
        if (lane == 0) {
            float m = mask ? mask[(size_t)s * BATCH + b] : 1.0f;
            sc[s] = (m > 0.0f) ? acc : NEGBIG;
        }
    }
    __syncthreads();

    float lm = -3.4e38f;
    for (int s = tid; s < S; s += 256) lm = fmaxf(lm, sc[s]);
    red[tid] = lm; __syncthreads();
    for (int o = 128; o > 0; o >>= 1) {
        if (tid < o) red[tid] = fmaxf(red[tid], red[tid + o]);
        __syncthreads();
    }
    float gmax = red[0]; __syncthreads();

    float ls = 0.0f;
    for (int s = tid; s < S; s += 256) ls += expf(sc[s] - gmax);
    red[tid] = ls; __syncthreads();
    for (int o = 128; o > 0; o >>= 1) {
        if (tid < o) red[tid] += red[tid + o];
        __syncthreads();
    }
    float gsum = red[0]; __syncthreads();

    for (int s = tid; s < S; s += 256) sc[s] = expf(sc[s] - gmax) / gsum;
    __syncthreads();

    for (int c = tid; c < CTXD; c += 256) {
        float acc = 0.0f;
        for (int s = 0; s < S; ++s)
            acc += sc[s] * ctx[((size_t)s * BATCH + b) * CTXD + c];
        z[(size_t)b * (2 * CTXD) + zoff + c] = acc;
    }
}

// =====================================================================
// Per-row log-softmax NLL over V; one block per batch row.
// =====================================================================
__global__ void vocab_loss(const float* __restrict__ logits, const int* __restrict__ ynext,
                           float* __restrict__ out)
{
    __shared__ float red[256];
    int b = blockIdx.x, tid = threadIdx.x;
    const float* row = logits + (size_t)b * NVOC;

    float lm = -3.4e38f;
    for (int i = tid; i < NVOC; i += 256) lm = fmaxf(lm, row[i]);
    red[tid] = lm; __syncthreads();
    for (int o = 128; o > 0; o >>= 1) {
        if (tid < o) red[tid] = fmaxf(red[tid], red[tid + o]);
        __syncthreads();
    }
    float gmax = red[0]; __syncthreads();

    float ls = 0.0f;
    for (int i = tid; i < NVOC; i += 256) ls += expf(row[i] - gmax);
    red[tid] = ls; __syncthreads();
    for (int o = 128; o > 0; o >>= 1) {
        if (tid < o) red[tid] += red[tid + o];
        __syncthreads();
    }
    if (tid == 0) {
        float lse = gmax + logf(red[0]);
        atomicAdd(out, lse - row[ynext[b]]);
    }
}

// =====================================================================
// Host launcher
// =====================================================================
static inline void gemm(const float* A, const float* W, const float* bias, float* D,
                        int M, int N, int K, int act, hipStream_t s)
{
    dim3 grid(N / 32, M / 16);
    gemm_f32_wmma<<<grid, 32, 0, s>>>(A, W, bias, D, M, N, K, act);
}

extern "C" void kernel_launch(void* const* d_in, const int* in_sizes, int n_in,
                              void* d_out, int out_size, void* d_ws, size_t ws_size,
                              hipStream_t stream)
{
    const float* txt_ctx   = (const float*)d_in[0];   // [STXT,B,CTX]
    const float* txt_mask  = (const float*)d_in[1];   // [STXT,B]
    const float* img_ctx   = (const float*)d_in[2];   // [SIMG,B,CTX]
    const int*   y         = (const int*)  d_in[3];   // [T,B]
    const float* emb_W     = (const float*)d_in[4];   // [V,E]
    const float* d0_Wih    = (const float*)d_in[5];
    const float* d0_Whh    = (const float*)d_in[6];
    const float* d0_bih    = (const float*)d_in[7];
    const float* d0_bhh    = (const float*)d_in[8];
    const float* d1_Wih    = (const float*)d_in[9];
    const float* d1_Whh    = (const float*)d_in[10];
    const float* d1_bih    = (const float*)d_in[11];
    const float* d1_bhh    = (const float*)d_in[12];
    const float* t_c2c     = (const float*)d_in[13];
    const float* t_h2c     = (const float*)d_in[14];
    const float* t_v       = (const float*)d_in[15];
    const float* i_c2c     = (const float*)d_in[16];
    const float* i_h2c     = (const float*)d_in[17];
    const float* i_v       = (const float*)d_in[18];
    const float* fusion_W  = (const float*)d_in[19];  // [H,2CTX]
    const float* h2o_W     = (const float*)d_in[20];  // [E,H]
    const float* h2o_b     = (const float*)d_in[21];
    const float* o2p_W     = (const float*)d_in[22];  // [V,E]
    const float* o2p_b     = (const float*)d_in[23];
    float* out = (float*)d_out;

    const int TM1  = TLEN - 1;               // 47 steps
    const int ROWS = TM1 * BATCH;            // 3008

    char* p = (char*)d_ws;
    auto take = [&](size_t bytes) { char* r = p; p += (bytes + 255) & ~(size_t)255; return r; };
    float*  ye      = (float*) take((size_t)ROWS * EMB * 4);
    float*  gi0     = (float*) take((size_t)ROWS * 3 * HID * 4);
    float*  tproj   = (float*) take((size_t)STXT * BATCH * CTXD * 4);
    float*  iproj   = (float*) take((size_t)SIMG * BATCH * CTXD * 4);
    __bf16* Wbf     = (__bf16*)take((size_t)NVOC * EMB * 2);
    float*  logitsv = (float*) take((size_t)BATCH * NVOC * 4);
    float*  Hbuf    = (float*) take((size_t)BATCH * HID * 4);
    float*  h1      = (float*) take((size_t)BATCH * HID * 4);
    float*  gh0     = (float*) take((size_t)BATCH * 3 * HID * 4);
    float*  gi1     = (float*) take((size_t)BATCH * 3 * HID * 4);
    float*  gh1     = (float*) take((size_t)BATCH * 3 * HID * 4);
    float*  hp_txt  = (float*) take((size_t)BATCH * CTXD * 4);
    float*  hp_img  = (float*) take((size_t)BATCH * CTXD * 4);
    float*  zcat    = (float*) take((size_t)BATCH * 2 * CTXD * 4);
    float*  fz      = (float*) take((size_t)BATCH * HID * 4);
    float*  logit   = (float*) take((size_t)BATCH * EMB * 4);
    __bf16* logitbf = (__bf16*)take((size_t)BATCH * EMB * 2);
    (void)ws_size; (void)in_sizes; (void)n_in; (void)out_size;

    // ---- init ----
    zero_f<<<1, 32, 0, stream>>>(out, 1);
    zero_f<<<(BATCH * HID + 255) / 256, 256, 0, stream>>>(Hbuf, BATCH * HID);

    // ---- hoisted time-invariant work ----
    f2bf_kernel<<<2048, 256, 0, stream>>>(o2p_W, Wbf, (long)NVOC * EMB);
    embed_gather<<<ROWS, 256, 0, stream>>>(y, emb_W, ye);
    gemm(ye, d0_Wih, d0_bih, gi0, ROWS, 3 * HID, EMB, 0, stream);
    gemm(txt_ctx, t_c2c, nullptr, tproj, STXT * BATCH, CTXD, CTXD, 0, stream);
    gemm(img_ctx, i_c2c, nullptr, iproj, SIMG * BATCH, CTXD, CTXD, 0, stream);

    // ---- sequential scan ----
    for (int t = 0; t < TM1; ++t) {
        const float* gi0_t = gi0 + (size_t)t * BATCH * 3 * HID;

        gemm(Hbuf, d0_Whh, d0_bhh, gh0, BATCH, 3 * HID, HID, 0, stream);
        gru_gates<<<(BATCH * HID + 255) / 256, 256, 0, stream>>>(gi0_t, gh0, Hbuf, h1);

        gemm(h1, t_h2c, nullptr, hp_txt, BATCH, CTXD, HID, 0, stream);
        gemm(h1, i_h2c, nullptr, hp_img, BATCH, CTXD, HID, 0, stream);
        att_kernel<<<BATCH, 256, 0, stream>>>(tproj, txt_ctx, hp_txt, t_v, txt_mask, zcat, STXT, 0);
        att_kernel<<<BATCH, 256, 0, stream>>>(iproj, img_ctx, hp_img, i_v, nullptr,  zcat, SIMG, CTXD);

        gemm(zcat, fusion_W, nullptr, fz, BATCH, HID, 2 * CTXD, 1, stream);
        gemm(fz, d1_Wih, d1_bih, gi1, BATCH, 3 * HID, HID, 0, stream);
        gemm(h1, d1_Whh, d1_bhh, gh1, BATCH, 3 * HID, HID, 0, stream);
        gru_gates<<<(BATCH * HID + 255) / 256, 256, 0, stream>>>(gi1, gh1, h1, Hbuf);

        gemm(Hbuf, h2o_W, h2o_b, logit, BATCH, EMB, HID, 1, stream);
        f2bf_kernel<<<64, 256, 0, stream>>>(logit, logitbf, (long)BATCH * EMB);

        gemm_bf16_vocab<<<NVOC / (16 * NT), 32, 0, stream>>>(logitbf, Wbf, o2p_b, logitsv);
        vocab_loss<<<BATCH, 256, 0, stream>>>(logitsv, y + (size_t)(t + 1) * BATCH, out);
    }
}
